// StateSpaceModule_71098888618418
// MI455X (gfx1250) — matrix operand order repaired and checked by reading, not compiled, not run on hardware
//
#include <hip/hip_runtime.h>
#include <math.h>

#define D 256
#define NLAYERS 2
#define NSLOTS 8
#define NHEADS 4
#define HDIM 64
#define BB 32
#define TT 1024

typedef __bf16 bf16_t;
typedef __attribute__((ext_vector_type(16))) __bf16 v16bf;
typedef __attribute__((ext_vector_type(8)))  float  v8f;

// ---------------- scalar <-> bf16 helpers (RNE) ----------------
__device__ __forceinline__ bf16_t f2bf(float f) {
  unsigned u = __builtin_bit_cast(unsigned, f);
  unsigned r = (u + 0x7FFFu + ((u >> 16) & 1u)) >> 16;
  return __builtin_bit_cast(bf16_t, (unsigned short)r);
}
__device__ __forceinline__ float bf2f(bf16_t h) {
  unsigned u = ((unsigned)__builtin_bit_cast(unsigned short, h)) << 16;
  return __builtin_bit_cast(float, u);
}

__device__ __forceinline__ v8f wmma_bf16(v16bf a, v16bf b, v8f c) {
  return __builtin_amdgcn_wmma_f32_16x16x32_bf16(false, a, false, b, (short)0, c,
                                                 false, false);
}

// B-fragment: pre-swizzled in global memory, frag (kt,nt) at
//   base + ((kt*NT + nt)*32 + lane)*16 bf16 elems  -> one contiguous 32B load/lane
__device__ __forceinline__ v16bf load_bfrag(const bf16_t* Wsw, int NT, int kt, int nt) {
  int lane = threadIdx.x & 31;
  const bf16_t* p = Wsw + (((size_t)(kt * NT + nt) * 32) + lane) * 16;
  return *(const v16bf*)p;
}

// prefetch address helper (frag base of a k-chunk for this wave)
__device__ __forceinline__ const void* Wsw_pf(const bf16_t* Wsw, int NT, int kt, int nt) {
  return (const void*)(Wsw + ((size_t)(kt * NT + nt) * 32) * 16);
}

// A-fragment (16x32 bf16) from LDS bf16, row stride lda
__device__ __forceinline__ v16bf load_afrag_lds(const bf16_t* As, int lda) {
  int lane = threadIdx.x & 31;
  int half = lane >> 4, m = lane & 15;
  v16bf a;
#pragma unroll
  for (int v = 0; v < 8; ++v) {
    int kb = (v < 4) ? (half * 8 + 2 * v) : (16 + half * 8 + 2 * (v - 4));
    a[2 * v]     = As[m * lda + kb];
    a[2 * v + 1] = As[m * lda + kb + 1];
  }
  return a;
}

// A-fragment from LDS f32 (converted on the fly)
__device__ __forceinline__ v16bf load_afrag_lds_f32(const float* As, int lda) {
  int lane = threadIdx.x & 31;
  int half = lane >> 4, m = lane & 15;
  v16bf a;
#pragma unroll
  for (int v = 0; v < 8; ++v) {
    int kb = (v < 4) ? (half * 8 + 2 * v) : (16 + half * 8 + 2 * (v - 4));
    a[2 * v]     = f2bf(As[m * lda + kb]);
    a[2 * v + 1] = f2bf(As[m * lda + kb + 1]);
  }
  return a;
}

__device__ __forceinline__ float sigmoid_f(float x) {
  return 1.0f / (1.0f + __expf(-x));
}

// ---------------- weight swizzle: f32 (N,K) row-major -> fragment-order bf16 ----------------
__global__ void k_swizzle_w(const float* __restrict__ W, bf16_t* __restrict__ out,
                            int K, int N) {
  int idx = blockIdx.x * blockDim.x + threadIdx.x;
  int total = K * N;
  if (idx >= total) return;
  int slot = idx & 15;
  int lane = (idx >> 4) & 31;
  int frag = idx >> 9;
  int NT = N >> 4;
  int kt = frag / NT, nt = frag - kt * NT;
  int k = kt * 32 + ((slot < 8) ? ((lane >> 4) * 8 + slot) : (16 + (lane >> 4) * 8 + (slot - 8)));
  int n = nt * 16 + (lane & 15);
  out[idx] = f2bf(W[(size_t)n * K + k]);
}

// ---------------- big tiled GEMM: Y = act( X @ W^T + bias ) ----------------
// X may be split across two f32 sources (concat along K). rowmap=1 gathers x as (B,T)->(T,B).
__global__ void k_gemm(const float* __restrict__ X0, const float* __restrict__ X1,
                       int K0, int K, int N,
                       const bf16_t* __restrict__ Wsw, const float* __restrict__ bias,
                       float* __restrict__ Y, int rowmap, int act) {
  __shared__ bf16_t As[64 * 32];
  int tid = threadIdx.x;                 // 256 threads = 8 waves
  int wave = tid >> 5, lane = tid & 31;
  int m0 = blockIdx.y * 64;
  int n0 = blockIdx.x * 128;
  int NT = N >> 4;
  int mt = wave >> 1;
  int nb = (wave & 1) * 4;
  const v8f vz = {0.f, 0.f, 0.f, 0.f, 0.f, 0.f, 0.f, 0.f};
  v8f acc[4] = {vz, vz, vz, vz};
  int K1 = K - K0;
  int kTiles = K >> 5;
  for (int kc = 0; kc < kTiles; ++kc) {
    for (int e = tid; e < 64 * 32; e += 256) {
      int row = e >> 5, kk = e & 31;
      int gr = m0 + row;
      int col = kc * 32 + kk;
      int sr = rowmap ? ((gr & (BB - 1)) * TT + (gr >> 5)) : gr;
      float xv = (col < K0) ? X0[(size_t)sr * K0 + col]
                            : X1[(size_t)sr * K1 + (col - K0)];
      As[e] = f2bf(xv);
    }
    __syncthreads();
    v16bf a = load_afrag_lds(As + mt * 16 * 32, 32);
#pragma unroll
    for (int j = 0; j < 4; ++j) {
      v16bf bfr = load_bfrag(Wsw, NT, kc, (n0 >> 4) + nb + j);
      acc[j] = wmma_bf16(a, bfr, acc[j]);
    }
    __syncthreads();
  }
  int half = lane >> 4, n = lane & 15;
#pragma unroll
  for (int j = 0; j < 4; ++j) {
    int dcol = n0 + (nb + j) * 16 + n;
    float bb = bias[dcol];
#pragma unroll
    for (int v = 0; v < 8; ++v) {
      int row = m0 + mt * 16 + half * 8 + v;
      float val = acc[j][v] + bb;
      if (act == 1) val = 0.5f * val * (1.0f + erff(val * 0.70710678f));
      Y[(size_t)row * N + dcol] = val;
    }
  }
}

// ---------------- persistent GRU scan (one workgroup, 512 thr = 16 waves) ----------------
__global__ void k_gru_scan(const float* __restrict__ xp,        // (T,B,3D)
                           const bf16_t* __restrict__ Whh_sw,   // swizzled (K=256,N=768)
                           const float* __restrict__ b_hh,      // (768)
                           float* __restrict__ seq_out,         // (T,B,D)
                           float* __restrict__ hp_g,            // scratch (32,768)
                           float* __restrict__ h_last_out) {    // (B,D)
  __shared__ float hs[BB * D];           // 32 KB
  int tid = threadIdx.x;
  for (int i = tid; i < BB * D; i += 512) hs[i] = 0.f;
  __syncthreads();
  int wave = tid >> 5, lane = tid & 31;
  int mt = wave & 1;
  int ng = wave >> 1;                    // 0..7 -> nt = ng*6 .. ng*6+5
  const int NT = 48;
  const v8f vz = {0.f, 0.f, 0.f, 0.f, 0.f, 0.f, 0.f, 0.f};
  for (int t = 0; t < TT; ++t) {
    v8f acc[6] = {vz, vz, vz, vz, vz, vz};
    for (int kc = 0; kc < 8; ++kc) {
      if (kc < 7)
        __builtin_prefetch(Wsw_pf(Whh_sw, NT, kc + 1, ng * 6), 0, 0);
      v16bf a = load_afrag_lds_f32(hs + (mt * 16) * D + kc * 32, D);
#pragma unroll
      for (int j = 0; j < 6; ++j) {
        v16bf bfr = load_bfrag(Whh_sw, NT, kc, ng * 6 + j);
        acc[j] = wmma_bf16(a, bfr, acc[j]);
      }
    }
    int half = lane >> 4, n = lane & 15;
#pragma unroll
    for (int j = 0; j < 6; ++j) {
      int col = (ng * 6 + j) * 16 + n;
      float bb = b_hh[col];
#pragma unroll
      for (int v = 0; v < 8; ++v) {
        int row = mt * 16 + half * 8 + v;
        hp_g[row * 768 + col] = acc[j][v] + bb;
      }
    }
    __threadfence();
    __syncthreads();
    const float* xpt = xp + (size_t)t * (BB * 768);
    for (int e = tid; e < BB * D; e += 512) {
      int b = e >> 8, d = e & 255;
      float xr = xpt[b * 768 + d];
      float xz = xpt[b * 768 + 256 + d];
      float xn = xpt[b * 768 + 512 + d];
      float hr = hp_g[b * 768 + d];
      float hz = hp_g[b * 768 + 256 + d];
      float hn = hp_g[b * 768 + 512 + d];
      float r = sigmoid_f(xr + hr);
      float z = sigmoid_f(xz + hz);
      float nn = tanhf(xn + r * hn);
      float h = hs[e];
      float hnew = (1.f - z) * nn + z * h;
      hs[e] = hnew;
      seq_out[(size_t)t * (BB * D) + e] = hnew;
    }
    __syncthreads();
  }
  for (int i = tid; i < BB * D; i += 512) h_last_out[i] = hs[i];
}

// ---------------- M=32 GEMM helper for the slow scan (all 16 waves) ----------------
template <typename Epi>
__device__ __forceinline__ void gemm32(const bf16_t* A0, const bf16_t* A1,
                                       const bf16_t* Wsw, int NT, int ntBase,
                                       int kTiles, const float* bias, Epi epi) {
  int tid = threadIdx.x, wave = tid >> 5, lane = tid & 31;
  int mt = wave & 1, ng = wave >> 1;     // nt = ng*2 + j
  const v8f vz = {0.f, 0.f, 0.f, 0.f, 0.f, 0.f, 0.f, 0.f};
  v8f acc[2] = {vz, vz};
  for (int kc = 0; kc < kTiles; ++kc) {
    const bf16_t* Asrc = (kc < 8) ? A0 : A1;
    int kcl = (kc < 8) ? kc : kc - 8;
    v16bf a = load_afrag_lds(Asrc + (mt * 16) * D + kcl * 32, D);
#pragma unroll
    for (int j = 0; j < 2; ++j) {
      v16bf bfr = load_bfrag(Wsw, NT, kc, ntBase + ng * 2 + j);
      acc[j] = wmma_bf16(a, bfr, acc[j]);
    }
  }
  int half = lane >> 4, n = lane & 15;
#pragma unroll
  for (int j = 0; j < 2; ++j) {
    int d = (ng * 2 + j) * 16 + n;
    float bb = bias ? bias[d] : 0.f;
#pragma unroll
    for (int v = 0; v < 8; ++v) {
      int brow = mt * 16 + half * 8 + v;
      epi(brow, d, acc[j][v] + bb, j, v);
    }
  }
}

// ---------------- slow-memory init ----------------
__global__ void k_slow_init_kv(const float* __restrict__ mem_slots,
                               const float* __restrict__ attn_in_w,
                               const float* __restrict__ attn_in_b,
                               float* __restrict__ k0, float* __restrict__ v0) {
  int i = blockIdx.x * blockDim.x + threadIdx.x;
  if (i >= NSLOTS * D) return;
  int s = i >> 8, d = i & 255;
  const float* wk = attn_in_w + (size_t)(256 + d) * D;
  const float* wv = attn_in_w + (size_t)(512 + d) * D;
  const float* ms = mem_slots + s * D;
  float ak = 0.f, av = 0.f;
  for (int k = 0; k < D; ++k) { ak += ms[k] * wk[k]; av += ms[k] * wv[k]; }
  k0[i] = ak + attn_in_b[256 + d];
  v0[i] = av + attn_in_b[512 + d];
}

__global__ void k_slow_broadcast(const float* __restrict__ mem_slots,
                                 const float* __restrict__ k0,
                                 const float* __restrict__ v0,
                                 float* __restrict__ memg,
                                 float* __restrict__ Kg, float* __restrict__ Vg) {
  int i = blockIdx.x * blockDim.x + threadIdx.x;
  if (i >= BB * NSLOTS * D) return;
  int sd = i & (NSLOTS * D - 1);
  memg[i] = mem_slots[sd];
  Kg[i] = k0[sd];
  Vg[i] = v0[sd];
}

// ---------------- persistent slow-memory scan (one workgroup, 512 thr) ----------------
__global__ void k_slow_scan(const float* __restrict__ fast_out,   // (T,B,D)
                            const bf16_t* __restrict__ attn_sw,   // (K=256,N=768) NT=48
                            const float* __restrict__ attn_in_b,
                            const bf16_t* __restrict__ attnout_sw,// NT=16
                            const float* __restrict__ attn_out_b,
                            const bf16_t* __restrict__ wg_sw,     // (K=512,N=256)
                            const float* __restrict__ wg_b,
                            const bf16_t* __restrict__ wp_sw,     // NT=16
                            const float* __restrict__ wp_b,
                            const bf16_t* __restrict__ outp_sw,   // (K=512,N=256)
                            const float* __restrict__ outp_b,
                            float* __restrict__ Kg, float* __restrict__ Vg,
                            float* __restrict__ memg, float* __restrict__ wcg,
                            float* __restrict__ enh, float* __restrict__ mem_out) {
  __shared__ bf16_t chunk_bf[BB * D];                 // 16 KB
  __shared__ bf16_t rp_bf[BB * D];                    // 16 KB
  __shared__ bf16_t aux_bf[BB * D];                   // 16 KB  (q -> read -> u)
  __shared__ float  sS[BB * NHEADS * NSLOTS];         // 4 KB
  __shared__ float  sWW[BB * NSLOTS];                 // 1 KB
  int tid = threadIdx.x;

  for (int t = 0; t < TT; ++t) {
    // P0: stage chunk
    const float* ck = fast_out + (size_t)t * (BB * D);
    for (int i = tid; i < BB * D; i += 512) chunk_bf[i] = f2bf(ck[i]);
    __syncthreads();

    float wcsave[2][8];
    // P1a: q = chunk @ Wq^T + bq  -> aux_bf
    gemm32(chunk_bf, (const bf16_t*)nullptr, attn_sw, 48, 0, 8, attn_in_b,
           [&](int b, int d, float val, int j, int v) {
             aux_bf[b * D + d] = f2bf(val);
           });
    // P1b: wc = chunk @ Wp^T + bp -> regs + global scratch
    gemm32(chunk_bf, (const bf16_t*)nullptr, wp_sw, 16, 0, 8, wp_b,
           [&](int b, int d, float val, int j, int v) {
             wcsave[j][v] = val;
             wcg[b * D + d] = val;
           });
    __threadfence();
    __syncthreads();

    // P2a: attention scores + softmax over slots
    for (int i = tid; i < BB * NHEADS * NSLOTS; i += 512) {
      int b = i >> 5, h = (i >> 3) & 3, s = i & 7;
      const float* kp = Kg + ((size_t)(b * NSLOTS + s)) * D + h * HDIM;
      const bf16_t* qp = aux_bf + b * D + h * HDIM;
      float acc = 0.f;
      for (int dd = 0; dd < HDIM; ++dd) acc += bf2f(qp[dd]) * kp[dd];
      sS[i] = acc * 0.125f;  // 1/sqrt(64)
    }
    __syncthreads();
    if (tid < BB * NHEADS) {
      float* row = sS + tid * NSLOTS;
      float mx = row[0];
      for (int s = 1; s < NSLOTS; ++s) mx = fmaxf(mx, row[s]);
      float sum = 0.f;
      for (int s = 0; s < NSLOTS; ++s) { row[s] = __expf(row[s] - mx); sum += row[s]; }
      float inv = 1.f / sum;
      for (int s = 0; s < NSLOTS; ++s) row[s] *= inv;
    }
    __syncthreads();
    // P2b: read = attn @ V  -> aux_bf (q fully consumed)
    for (int i = tid; i < BB * D; i += 512) {
      int b = i >> 8, d = i & 255, h = d >> 6;
      const float* arow = sS + (b * NHEADS + h) * NSLOTS;
      float r = 0.f;
#pragma unroll
      for (int s = 0; s < NSLOTS; ++s)
        r += arow[s] * Vg[((size_t)(b * NSLOTS + s)) * D + d];
      aux_bf[i] = f2bf(r);
    }
    __syncthreads();

    // P3: read_proj -> rp_bf
    gemm32(aux_bf, (const bf16_t*)nullptr, attnout_sw, 16, 0, 8, attn_out_b,
           [&](int b, int d, float val, int j, int v) {
             rp_bf[b * D + d] = f2bf(val);
           });
    __syncthreads();

    // P4: enhanced = [chunk|rp] @ outp^T + b -> enh global
    float* enht = enh + (size_t)t * (BB * D);
    gemm32(chunk_bf, rp_bf, outp_sw, 16, 0, 16, outp_b,
           [&](int b, int d, float val, int j, int v) {
             enht[b * D + d] = val;
           });

    // P4.5: ww = softmax(wc . mem / sqrt(D)) over slots (pre-update mem)
    if (tid < BB * NSLOTS) {
      int b = tid >> 3, s = tid & 7;
      const float* wcp = wcg + b * D;
      const float* mp = memg + ((size_t)(b * NSLOTS + s)) * D;
      float acc = 0.f;
      for (int k = 0; k < D; ++k) acc += wcp[k] * mp[k];
      sWW[tid] = acc * 0.0625f;  // 1/sqrt(256)
    }
    __syncthreads();
    if (tid < BB) {
      float* row = sWW + tid * NSLOTS;
      float mx = row[0];
      for (int s = 1; s < NSLOTS; ++s) mx = fmaxf(mx, row[s]);
      float sum = 0.f;
      for (int s = 0; s < NSLOTS; ++s) { row[s] = __expf(row[s] - mx); sum += row[s]; }
      float inv = 1.f / sum;
      for (int s = 0; s < NSLOTS; ++s) row[s] *= inv;
    }
    __syncthreads();

    // P5: gate = sigmoid([chunk|rp] @ wg^T + b); u = gate*wc; mem += ww (x) u
    gemm32(chunk_bf, rp_bf, wg_sw, 16, 0, 16, wg_b,
           [&](int b, int d, float val, int j, int v) {
             float u = sigmoid_f(val) * wcsave[j][v];
             aux_bf[b * D + d] = f2bf(u);
             float* mp = memg + (size_t)b * NSLOTS * D + d;
             const float* wwp = sWW + b * NSLOTS;
#pragma unroll
             for (int s = 0; s < NSLOTS; ++s) mp[s * D] += wwp[s] * u;
           });
    __syncthreads();

    // P7: incremental K/V rank-1 updates:  K += ww (x) (u @ Wk^T),  V += ww (x) (u @ Wv^T)
    gemm32(aux_bf, (const bf16_t*)nullptr, attn_sw, 48, 16, 8, (const float*)nullptr,
           [&](int b, int d, float val, int j, int v) {
             float* kp = Kg + (size_t)b * NSLOTS * D + d;
             const float* wwp = sWW + b * NSLOTS;
#pragma unroll
             for (int s = 0; s < NSLOTS; ++s) kp[s * D] += wwp[s] * val;
           });
    gemm32(aux_bf, (const bf16_t*)nullptr, attn_sw, 48, 32, 8, (const float*)nullptr,
           [&](int b, int d, float val, int j, int v) {
             float* vp = Vg + (size_t)b * NSLOTS * D + d;
             const float* wwp = sWW + b * NSLOTS;
#pragma unroll
             for (int s = 0; s < NSLOTS; ++s) vp[s * D] += wwp[s] * val;
           });
    __threadfence();
    __syncthreads();
  }
  for (int i = tid; i < BB * NSLOTS * D; i += 512) mem_out[i] = memg[i];
}

// ---------------- residual + LayerNorm (one wave32 per row) ----------------
__global__ void k_ln_residual(const float* __restrict__ a, const float* __restrict__ res,
                              int res_map, const float* __restrict__ g,
                              const float* __restrict__ bta, float* __restrict__ out,
                              int out_map) {
  int wave = threadIdx.x >> 5, lane = threadIdx.x & 31;
  int r = blockIdx.x * 8 + wave;         // r = t*B + b
  int t = r >> 5, b = r & (BB - 1);
  size_t ra = (size_t)r * D;
  size_t rres = res_map ? ((size_t)(b * TT + t) * D) : ra;
  float vals[8];
  float sum = 0.f;
#pragma unroll
  for (int i = 0; i < 8; ++i) {
    int d = lane + i * 32;
    float x = a[ra + d] + res[rres + d];
    vals[i] = x;
    sum += x;
  }
#pragma unroll
  for (int off = 16; off > 0; off >>= 1) sum += __shfl_xor(sum, off, 32);
  float mean = sum * (1.f / 256.f);
  float var = 0.f;
#pragma unroll
  for (int i = 0; i < 8; ++i) {
    float dv = vals[i] - mean;
    var += dv * dv;
  }
#pragma unroll
  for (int off = 16; off > 0; off >>= 1) var += __shfl_xor(var, off, 32);
  var *= (1.f / 256.f);
  float inv = rsqrtf(var + 1e-5f);
  size_t rout = out_map ? ((size_t)(b * TT + t) * D) : ra;
#pragma unroll
  for (int i = 0; i < 8; ++i) {
    int d = lane + i * 32;
    out[rout + d] = (vals[i] - mean) * inv * g[d] + bta[d];
  }
}

// ---------------- host orchestration ----------------
extern "C" void kernel_launch(void* const* d_in, const int* in_sizes, int n_in,
                              void* d_out, int out_size, void* d_ws, size_t ws_size,
                              hipStream_t stream) {
  (void)in_sizes; (void)n_in; (void)out_size; (void)ws_size;
  const float* x          = (const float*)d_in[0];
  const float* gru_w_ih   = (const float*)d_in[1];
  const float* gru_w_hh   = (const float*)d_in[2];
  const float* gru_b_ih   = (const float*)d_in[3];
  const float* gru_b_hh   = (const float*)d_in[4];
  const float* fast_ln_g  = (const float*)d_in[5];
  const float* fast_ln_b  = (const float*)d_in[6];
  const float* mem_slots  = (const float*)d_in[7];
  const float* attn_in_w  = (const float*)d_in[8];
  const float* attn_in_b  = (const float*)d_in[9];
  const float* attn_out_w = (const float*)d_in[10];
  const float* attn_out_b = (const float*)d_in[11];
  const float* wg_w       = (const float*)d_in[12];
  const float* wg_b       = (const float*)d_in[13];
  const float* wp_w       = (const float*)d_in[14];
  const float* wp_b       = (const float*)d_in[15];
  const float* outp_w     = (const float*)d_in[16];
  const float* outp_b     = (const float*)d_in[17];
  const float* slow_ln_g  = (const float*)d_in[18];
  const float* slow_ln_b  = (const float*)d_in[19];
  const float* fus1_w     = (const float*)d_in[20];
  const float* fus1_b     = (const float*)d_in[21];
  const float* fus2_w     = (const float*)d_in[22];
  const float* fus2_b     = (const float*)d_in[23];
  const float* final_ln_g = (const float*)d_in[24];
  const float* final_ln_b = (const float*)d_in[25];

  float* out_main   = (float*)d_out;                          // (B,T,D)
  float* out_hidden = out_main + (size_t)BB * TT * D;         // (L,B,D)
  float* out_mem    = out_hidden + (size_t)NLAYERS * BB * D;  // (B,S,D)

  char* wsp = (char*)d_ws;
  auto alloc = [&](size_t bytes) -> char* {
    char* p = wsp;
    wsp += (bytes + 255) & ~(size_t)255;
    return p;
  };
  bf16_t* sw_ih0   = (bf16_t*)alloc((size_t)768 * 256 * 2);
  bf16_t* sw_ih1   = (bf16_t*)alloc((size_t)768 * 256 * 2);
  bf16_t* sw_hh0   = (bf16_t*)alloc((size_t)768 * 256 * 2);
  bf16_t* sw_hh1   = (bf16_t*)alloc((size_t)768 * 256 * 2);
  bf16_t* sw_attn  = (bf16_t*)alloc((size_t)768 * 256 * 2);
  bf16_t* sw_aout  = (bf16_t*)alloc((size_t)256 * 256 * 2);
  bf16_t* sw_wp    = (bf16_t*)alloc((size_t)256 * 256 * 2);
  bf16_t* sw_outp  = (bf16_t*)alloc((size_t)256 * 512 * 2);
  bf16_t* sw_wg    = (bf16_t*)alloc((size_t)256 * 512 * 2);
  bf16_t* sw_fus1  = (bf16_t*)alloc((size_t)256 * 512 * 2);
  bf16_t* sw_fus2  = (bf16_t*)alloc((size_t)256 * 256 * 2);
  float* hp_g  = (float*)alloc((size_t)BB * 768 * 4);
  float* wcg   = (float*)alloc((size_t)BB * D * 4);
  float* k0    = (float*)alloc((size_t)NSLOTS * D * 4);
  float* v0    = (float*)alloc((size_t)NSLOTS * D * 4);
  float* Kg    = (float*)alloc((size_t)BB * NSLOTS * D * 4);
  float* Vg    = (float*)alloc((size_t)BB * NSLOTS * D * 4);
  float* memg  = (float*)alloc((size_t)BB * NSLOTS * D * 4);
  float* xp    = (float*)alloc((size_t)TT * BB * 768 * 4);    // 96 MB (reused)
  float* seq1  = (float*)alloc((size_t)TT * BB * D * 4);      // 32 MB
  float* seq2  = (float*)alloc((size_t)TT * BB * D * 4);      // 32 MB
  float* fasto = (float*)alloc((size_t)TT * BB * D * 4);      // 32 MB
  // overlays (xp dead after GRU; seq1/seq2 dead after fast_out)
  float* enh      = xp;
  float* slow_out = xp + (size_t)TT * BB * D;
  float* g1 = seq1;
  float* f2 = seq2;

  // --- weight swizzles ---
  k_swizzle_w<<<768, 256, 0, stream>>>(gru_w_ih, sw_ih0, 256, 768);
  k_swizzle_w<<<768, 256, 0, stream>>>(gru_w_ih + (size_t)768 * 256, sw_ih1, 256, 768);
  k_swizzle_w<<<768, 256, 0, stream>>>(gru_w_hh, sw_hh0, 256, 768);
  k_swizzle_w<<<768, 256, 0, stream>>>(gru_w_hh + (size_t)768 * 256, sw_hh1, 256, 768);
  k_swizzle_w<<<768, 256, 0, stream>>>(attn_in_w, sw_attn, 256, 768);
  k_swizzle_w<<<256, 256, 0, stream>>>(attn_out_w, sw_aout, 256, 256);
  k_swizzle_w<<<256, 256, 0, stream>>>(wp_w, sw_wp, 256, 256);
  k_swizzle_w<<<512, 256, 0, stream>>>(outp_w, sw_outp, 512, 256);
  k_swizzle_w<<<512, 256, 0, stream>>>(wg_w, sw_wg, 512, 256);
  k_swizzle_w<<<512, 256, 0, stream>>>(fus1_w, sw_fus1, 512, 256);
  k_swizzle_w<<<256, 256, 0, stream>>>(fus2_w, sw_fus2, 256, 256);

  dim3 blk(256);
  // --- GRU layer 0 ---
  k_gemm<<<dim3(6, 512), blk, 0, stream>>>(x, x, 256, 256, 768, sw_ih0, gru_b_ih,
                                           xp, /*rowmap=*/1, /*act=*/0);
  k_gru_scan<<<1, 512, 0, stream>>>(xp, sw_hh0, gru_b_hh, seq1, hp_g, out_hidden);
  // --- GRU layer 1 ---
  k_gemm<<<dim3(6, 512), blk, 0, stream>>>(seq1, seq1, 256, 256, 768, sw_ih1,
                                           gru_b_ih + 768, xp, 0, 0);
  k_gru_scan<<<1, 512, 0, stream>>>(xp, sw_hh1, gru_b_hh + 768, seq2, hp_g,
                                    out_hidden + BB * D);
  // --- fast_out = LN(seq2 + x) (kept in (T,B,D)) ---
  k_ln_residual<<<TT * BB / 8, 256, 0, stream>>>(seq2, x, 1, fast_ln_g, fast_ln_b,
                                                 fasto, 0);
  // --- slow memory ---
  k_slow_init_kv<<<8, 256, 0, stream>>>(mem_slots, attn_in_w, attn_in_b, k0, v0);
  k_slow_broadcast<<<256, 256, 0, stream>>>(mem_slots, k0, v0, memg, Kg, Vg);
  k_slow_scan<<<1, 512, 0, stream>>>(fasto, sw_attn, attn_in_b, sw_aout, attn_out_b,
                                     sw_wg, wg_b, sw_wp, wp_b, sw_outp, outp_b,
                                     Kg, Vg, memg, wcg, enh, out_mem);
  k_ln_residual<<<TT * BB / 8, 256, 0, stream>>>(enh, fasto, 0, slow_ln_g, slow_ln_b,
                                                 slow_out, 0);
  // --- fusion MLP + final LN (output transposed back to (B,T,D)) ---
  k_gemm<<<dim3(2, 512), blk, 0, stream>>>(fasto, slow_out, 256, 512, 256, sw_fus1,
                                           fus1_b, g1, 0, /*gelu=*/1);
  k_gemm<<<dim3(2, 512), blk, 0, stream>>>(g1, g1, 256, 256, 256, sw_fus2, fus2_b,
                                           f2, 0, 0);
  k_ln_residual<<<TT * BB / 8, 256, 0, stream>>>(f2, x, 1, final_ln_g, final_ln_b,
                                                 out_main, 1);
}
